// Self_Attn_20023137534817
// MI455X (gfx1250) — matrix operand order repaired and checked by reading, not compiled, not run on hardware
//
#include <hip/hip_runtime.h>

// ---------------------------------------------------------------------------
// Types for CDNA5 WMMA (wave32): v_wmma_f32_16x16x32_bf16
// ---------------------------------------------------------------------------
typedef __attribute__((ext_vector_type(16))) __bf16 bf16x16;
typedef __attribute__((ext_vector_type(8)))  __bf16 bf16x8;
typedef __attribute__((ext_vector_type(2)))  __bf16 bf16x2;
typedef __attribute__((ext_vector_type(8)))  float  f32x8;

// Build a 16-element bf16 fragment from two 16-byte LDS chunks (ds_load_b128).
static __device__ __forceinline__ bf16x16 frag16(const __bf16* p0, const __bf16* p1) {
  bf16x8 lo = *reinterpret_cast<const bf16x8*>(p0);
  bf16x8 hi = *reinterpret_cast<const bf16x8*>(p1);
  return __builtin_shufflevector(lo, hi, 0, 1, 2, 3, 4, 5, 6, 7,
                                 8, 9, 10, 11, 12, 13, 14, 15);
}

static __device__ __forceinline__ f32x8 wmma_bf16(bf16x16 a, bf16x16 b, f32x8 c) {
  return __builtin_amdgcn_wmma_f32_16x16x32_bf16(false, a, false, b,
                                                 (short)0, c, false, false);
}

// Async DMA: global -> LDS, 16 bytes per lane (GVS mode: SADDR base + VGPR i32
// offset).  Tracked by ASYNCcnt.  (CDNA5 ISA)
static __device__ __forceinline__ void async16(unsigned lds_off, const void* sbase,
                                               unsigned goff) {
  asm volatile("global_load_async_to_lds_b128 %0, %1, %2"
               :: "v"(lds_off), "v"(goff), "s"(sbase) : "memory");
}
static __device__ __forceinline__ void wait_async0() {
  asm volatile("s_wait_asynccnt 0x0" ::: "memory");
}
static __device__ __forceinline__ void wait_async4() {   // keep 1 batch in flight
  asm volatile("s_wait_asynccnt 0x4" ::: "memory");
}
static __device__ __forceinline__ unsigned lds_addr(const void* p) {
  return (unsigned)(uintptr_t)p;   // low 32 bits of generic ptr = LDS byte addr
}

#define ZERO8 {0.f, 0.f, 0.f, 0.f, 0.f, 0.f, 0.f, 0.f}

#define BATCH 16
#define CH    256
#define MED   128
#define NPIX  4096

// ---------------------------------------------------------------------------
// Kernel 0: one-time fp32 -> bf16 conversion of the three weight matrices.
// grid 128, block 256 (each array is 128*256 = 32768 elements).
// ---------------------------------------------------------------------------
__global__ __launch_bounds__(256) void k_prep_weights(
    const float* __restrict__ w_o, const float* __restrict__ w_v,
    const float* __restrict__ w_c, __bf16* __restrict__ wo_bf,
    __bf16* __restrict__ wv_bf, __bf16* __restrict__ wc_bf)
{
  const int i = blockIdx.x * 256 + threadIdx.x;
  wo_bf[i] = (__bf16)w_o[i];
  wv_bf[i] = (__bf16)w_v[i];
  wc_bf[i] = (__bf16)w_c[i];
}

// ---------------------------------------------------------------------------
// Kernel 1: projections  o = w_o*x + b_o (bf16 out),  v = w_v*x + b_v (fp32).
// grid (NPIX/64, BATCH), block 256.  Weight tiles DMA'd with async-to-LDS
// (layout [m][32] == global rows); x tile converted + transposed manually
// with packed bf16x2 (ds_store_b32) while the DMA is in flight.
// ---------------------------------------------------------------------------
__global__ __launch_bounds__(256) void k_proj(
    const float* __restrict__ x,
    const __bf16* __restrict__ wo_bf, const __bf16* __restrict__ wv_bf,
    const float* __restrict__ b_o, const float* __restrict__ b_v,
    __bf16* __restrict__ o_bf, float* __restrict__ v_out)
{
  __shared__ __align__(16) __bf16 sWo[128 * 32];
  __shared__ __align__(16) __bf16 sWv[128 * 32];
  __shared__ __align__(16) __bf16 sX[64 * 32];

  const int b    = blockIdx.y;
  const int n0   = blockIdx.x * 64;
  const int tid  = threadIdx.x;
  const int wave = tid >> 5;
  const int lane = tid & 31;
  const int half = (lane >> 4) & 1;
  const int ln   = lane & 15;
  const int mb   = wave * 16;

  const unsigned lWo = lds_addr(sWo), lWv = lds_addr(sWv);

  f32x8 acc_o[4] = {ZERO8, ZERO8, ZERO8, ZERO8};
  f32x8 acc_v[4] = {ZERO8, ZERO8, ZERO8, ZERO8};

  for (int k0 = 0; k0 < CH; k0 += 32) {
    // async DMA: weight tiles [128][32] bf16 (512 16B chunks each)
    for (int h = 0; h < 2; ++h) {
      const int q = tid + h * 256;            // 0..511
      const int r = q >> 2, ch = q & 3;
      const unsigned goff = (unsigned)(r * CH + k0) * 2 + ch * 16;
      async16(lWo + r * 64 + ch * 16, wo_bf, goff);
      async16(lWv + r * 64 + ch * 16, wv_bf, goff);
    }
    // x tile: 32 k x 64 n -> [n][k] bf16, packed pairs over k
    for (int idx = tid; idx < 16 * 64; idx += 256) {
      const int kp = idx >> 6, n = idx & 63;
      bf16x2 p;
      p.x = (__bf16)x[((size_t)b * CH + k0 + 2 * kp) * NPIX + n0 + n];
      p.y = (__bf16)x[((size_t)b * CH + k0 + 2 * kp + 1) * NPIX + n0 + n];
      *reinterpret_cast<bf16x2*>(&sX[n * 32 + 2 * kp]) = p;
    }
    wait_async0();
    __syncthreads();

    const int ra = (mb + ln) * 32;
    bf16x16 ao = frag16(&sWo[ra + half * 8], &sWo[ra + 16 + half * 8]);
    bf16x16 av = frag16(&sWv[ra + half * 8], &sWv[ra + 16 + half * 8]);
    for (int t = 0; t < 4; ++t) {
      const int rb = (t * 16 + ln) * 32;
      bf16x16 bx = frag16(&sX[rb + half * 8], &sX[rb + 16 + half * 8]);
      acc_o[t] = wmma_bf16(ao, bx, acc_o[t]);
      acc_v[t] = wmma_bf16(av, bx, acc_v[t]);
    }
    __syncthreads();
  }

  for (int t = 0; t < 4; ++t) {
    const int n = n0 + t * 16 + ln;
    for (int r = 0; r < 8; ++r) {
      const int m = mb + r + half * 8;
      o_bf[((size_t)b * MED + m) * NPIX + n]  = (__bf16)(acc_o[t][r] + b_o[m]);
      v_out[((size_t)b * MED + m) * NPIX + n] = acc_v[t][r] + b_v[m];
    }
  }
}

// ---------------------------------------------------------------------------
// Kernel 2: column softmax over M=128 (x_s branch).  v fp32 -> xs bf16.
// ---------------------------------------------------------------------------
__global__ __launch_bounds__(256) void k_softmax_colM(
    const float* __restrict__ v, __bf16* __restrict__ xs)
{
  const int b = blockIdx.y;
  const int n = blockIdx.x * 256 + threadIdx.x;
  const size_t base = (size_t)b * MED * NPIX + n;

  float mx = -3.4e38f;
  for (int m = 0; m < MED; ++m)
    mx = fmaxf(mx, v[base + (size_t)m * NPIX]);
  float sum = 0.f;
  for (int m = 0; m < MED; ++m)
    sum += __expf(v[base + (size_t)m * NPIX] - mx);
  const float inv = 1.f / sum;
  for (int m = 0; m < MED; ++m)
    xs[base + (size_t)m * NPIX] =
        (__bf16)(__expf(v[base + (size_t)m * NPIX] - mx) * inv);
}

// ---------------------------------------------------------------------------
// Kernel 3: row softmax over N=4096 (x_v branch).  v fp32 -> xv bf16.
// ---------------------------------------------------------------------------
__global__ __launch_bounds__(256) void k_softmax_rowN(
    const float* __restrict__ v, __bf16* __restrict__ xv)
{
  __shared__ float red[256];
  const int tid = threadIdx.x;
  const size_t base = (size_t)blockIdx.x * NPIX;

  float vals[16];
  float mx = -3.4e38f;
  for (int j = 0; j < 16; ++j) {
    vals[j] = v[base + tid + j * 256];
    mx = fmaxf(mx, vals[j]);
  }
  red[tid] = mx;
  __syncthreads();
  for (int s = 128; s > 0; s >>= 1) {
    if (tid < s) red[tid] = fmaxf(red[tid], red[tid + s]);
    __syncthreads();
  }
  mx = red[0];
  __syncthreads();

  float sum = 0.f;
  for (int j = 0; j < 16; ++j) {
    vals[j] = __expf(vals[j] - mx);
    sum += vals[j];
  }
  red[tid] = sum;
  __syncthreads();
  for (int s = 128; s > 0; s >>= 1) {
    if (tid < s) red[tid] += red[tid + s];
    __syncthreads();
  }
  const float inv = 1.f / red[0];
  for (int j = 0; j < 16; ++j)
    xv[base + tid + j * 256] = (__bf16)(vals[j] * inv);
}

// ---------------------------------------------------------------------------
// Kernel 4: split-K Gram partials  A[m,k] = sum_n xv[m,n]*xs[k,n].
// grid (16 slices, BATCH), block 256.  Operand tiles are bit-copies of
// global bf16 rows -> double-buffered async DMA, one batch kept in flight.
// ---------------------------------------------------------------------------
__global__ __launch_bounds__(256) void k_gram_partial(
    const __bf16* __restrict__ xv, const __bf16* __restrict__ xs,
    float* __restrict__ Apart)
{
  __shared__ __align__(16) __bf16 sA[2][128 * 32];
  __shared__ __align__(16) __bf16 sB[2][128 * 32];

  const int ks   = blockIdx.x;
  const int b    = blockIdx.y;
  const int tid  = threadIdx.x;
  const int wave = tid >> 5;
  const int lane = tid & 31;
  const int half = (lane >> 4) & 1;
  const int ln   = lane & 15;
  const int mb   = wave * 16;
  const int nbeg = ks * 256;

  const __bf16* gA = xv + (size_t)b * MED * NPIX;
  const __bf16* gB = xs + (size_t)b * MED * NPIX;
  const unsigned la[2] = {lds_addr(sA[0]), lds_addr(sA[1])};
  const unsigned lb[2] = {lds_addr(sB[0]), lds_addr(sB[1])};

  f32x8 acc[8] = {ZERO8, ZERO8, ZERO8, ZERO8, ZERO8, ZERO8, ZERO8, ZERO8};

  auto issue = [&](int buf, int n0) {   // 4 async instr / thread (2 per tile)
    for (int h = 0; h < 2; ++h) {
      const int q = tid + h * 256;      // 0..511
      const int r = q >> 2, ch = q & 3;
      const unsigned goff = (unsigned)(r * NPIX + n0) * 2 + ch * 16;
      async16(la[buf] + r * 64 + ch * 16, gA, goff);
      async16(lb[buf] + r * 64 + ch * 16, gB, goff);
    }
  };

  issue(0, nbeg);
  for (int s = 0; s < 8; ++s) {
    const int cur = s & 1;
    if (s < 7) { issue(1 - cur, nbeg + (s + 1) * 32); wait_async4(); }
    else       { wait_async0(); }
    __syncthreads();

    const int ra = (mb + ln) * 32;
    bf16x16 af = frag16(&sA[cur][ra + half * 8], &sA[cur][ra + 16 + half * 8]);
    for (int j = 0; j < 8; ++j) {
      const int rb = (j * 16 + ln) * 32;
      bf16x16 bf = frag16(&sB[cur][rb + half * 8], &sB[cur][rb + 16 + half * 8]);
      acc[j] = wmma_bf16(af, bf, acc[j]);
    }
    __syncthreads();
  }

  float* out = Apart + ((size_t)ks * BATCH + b) * (MED * MED);
  for (int j = 0; j < 8; ++j) {
    const int kc = j * 16 + ln;
    for (int r = 0; r < 8; ++r)
      out[(mb + r + half * 8) * MED + kc] = acc[j][r];
  }
}

// Kernel 5: reduce 16 split-K partials -> A.  grid 1024, block 256.
__global__ __launch_bounds__(256) void k_reduceA(
    const float* __restrict__ Apart, float* __restrict__ A)
{
  const size_t i = (size_t)blockIdx.x * 256 + threadIdx.x;
  const size_t b = i >> 14, e = i & 16383;
  float s = 0.f;
  for (int ks = 0; ks < 16; ++ks)
    s += Apart[((size_t)ks * BATCH + b) * (MED * MED) + e];
  A[i] = s;
}

// ---------------------------------------------------------------------------
// Kernel 6: fused tail.  Bt = A^T * o, then out = relu(w_c*Bt + b_c),
// result = gamma*out + x.  64 KB static LDS:
//   [0,32K)   sAt (A^T bf16 [k][m])  -> reused as 2 x 16K w_c buffers
//   [32K,48K) sO  (o tile bf16 [n][m])
//   [48K,64K) sBt (Bt bf16 [n][k])
// ---------------------------------------------------------------------------
__global__ __launch_bounds__(256) void k_final(
    const __bf16* __restrict__ o_bf, const float* __restrict__ Ain,
    const __bf16* __restrict__ wc_bf, const float* __restrict__ b_c,
    const float* __restrict__ gamma, const float* __restrict__ x,
    float* __restrict__ out)
{
  __shared__ __align__(16) char smem[65536];
  __bf16* sAt = (__bf16*)smem;
  __bf16* sO  = (__bf16*)(smem + 32768);
  __bf16* sBt = (__bf16*)(smem + 49152);

  const int b    = blockIdx.y;
  const int n0   = blockIdx.x * 64;
  const int tid  = threadIdx.x;
  const int wave = tid >> 5;
  const int lane = tid & 31;
  const int half = (lane >> 4) & 1;
  const int ln   = lane & 15;

  // Stage A^T (fp32 -> bf16) and o tile, both with packed bf16x2 stores.
  for (int idx = tid; idx < 64 * 128; idx += 256) {
    const int mp = idx >> 7, k = idx & 127;
    bf16x2 p;
    p.x = (__bf16)Ain[(size_t)b * 16384 + (2 * mp) * 128 + k];
    p.y = (__bf16)Ain[(size_t)b * 16384 + (2 * mp + 1) * 128 + k];
    *reinterpret_cast<bf16x2*>(&sAt[k * 128 + 2 * mp]) = p;
  }
  for (int idx = tid; idx < 64 * 64; idx += 256) {
    const int mp = idx >> 6, n = idx & 63;
    bf16x2 p;
    p.x = o_bf[((size_t)b * MED + 2 * mp) * NPIX + n0 + n];
    p.y = o_bf[((size_t)b * MED + 2 * mp + 1) * NPIX + n0 + n];
    *reinterpret_cast<bf16x2*>(&sO[n * 128 + 2 * mp]) = p;
  }
  __syncthreads();

  // GEMM1: Bt[k,n] = sum_m At[k,m]*O[m,n]; wave owns k-strip wave*16.
  {
    const int kb = wave * 16;
    f32x8 acc[4] = {ZERO8, ZERO8, ZERO8, ZERO8};
    for (int s = 0; s < 4; ++s) {
      const int ra = (kb + ln) * 128 + s * 32;
      bf16x16 af = frag16(&sAt[ra + half * 8], &sAt[ra + 16 + half * 8]);
      for (int t = 0; t < 4; ++t) {
        const int rb = (t * 16 + ln) * 128 + s * 32;
        bf16x16 bf = frag16(&sO[rb + half * 8], &sO[rb + 16 + half * 8]);
        acc[t] = wmma_bf16(af, bf, acc[t]);
      }
    }
    // transpose into sBt[n][k]; 8 rows are k-contiguous -> one b128 store
    for (int t = 0; t < 4; ++t) {
      const int n = t * 16 + ln;
      bf16x8 pk;
      for (int r = 0; r < 8; ++r) pk[r] = (__bf16)acc[t][r];
      *reinterpret_cast<bf16x8*>(&sBt[n * 128 + kb + half * 8]) = pk;
    }
  }
  __syncthreads();   // sAt retired; its space becomes the w_c double buffer

  // GEMM2: out[c,n] = w_c[c,:]*Bt[:,n]; async double-buffered w_c tiles.
  f32x8 acc2[2][4] = {{ZERO8, ZERO8, ZERO8, ZERO8},
                      {ZERO8, ZERO8, ZERO8, ZERO8}};
  const int cb[2] = {wave * 16, 128 + wave * 16};
  const unsigned lWc = lds_addr(smem);

  auto issueW = [&](int buf, int s) {   // 256x32 bf16 tile = 1024 16B chunks
    const unsigned lbase = lWc + (unsigned)buf * 16384u;
    for (int h = 0; h < 4; ++h) {
      const int q = tid + h * 256;
      const int c = q >> 2, ch = q & 3;
      async16(lbase + c * 64 + ch * 16, wc_bf,
              (unsigned)(c * MED + s * 32) * 2 + ch * 16);
    }
  };

  issueW(0, 0);
  for (int s = 0; s < 4; ++s) {
    const int cur = s & 1;
    if (s < 3) { issueW(1 - cur, s + 1); wait_async4(); }
    else       { wait_async0(); }
    __syncthreads();

    const __bf16* sWc = (const __bf16*)smem + cur * 8192;
    const int r0 = (cb[0] + ln) * 32;
    const int r1 = (cb[1] + ln) * 32;
    bf16x16 a0 = frag16(&sWc[r0 + half * 8], &sWc[r0 + 16 + half * 8]);
    bf16x16 a1 = frag16(&sWc[r1 + half * 8], &sWc[r1 + 16 + half * 8]);
    for (int t = 0; t < 4; ++t) {
      const int rb = (t * 16 + ln) * 128 + s * 32;
      bf16x16 bf = frag16(&sBt[rb + half * 8], &sBt[rb + 16 + half * 8]);
      acc2[0][t] = wmma_bf16(a0, bf, acc2[0][t]);
      acc2[1][t] = wmma_bf16(a1, bf, acc2[1][t]);
    }
    __syncthreads();
  }

  const float g = gamma[0];
  for (int u = 0; u < 2; ++u) {
    for (int t = 0; t < 4; ++t) {
      const int n = n0 + t * 16 + ln;
      for (int r = 0; r < 8; ++r) {
        const int c = cb[u] + r + half * 8;
        const float val = fmaxf(acc2[u][t][r] + b_c[c], 0.f);
        const size_t gi = ((size_t)b * CH + c) * NPIX + n;
        out[gi] = g * val + x[gi];
      }
    }
  }
}

// ---------------------------------------------------------------------------
extern "C" void kernel_launch(void* const* d_in, const int* in_sizes, int n_in,
                              void* d_out, int out_size, void* d_ws,
                              size_t ws_size, hipStream_t stream)
{
  const float* x     = (const float*)d_in[0];
  const float* w_o   = (const float*)d_in[1];
  const float* b_o   = (const float*)d_in[2];
  const float* w_v   = (const float*)d_in[3];
  const float* b_v   = (const float*)d_in[4];
  const float* w_c   = (const float*)d_in[5];
  const float* b_c   = (const float*)d_in[6];
  const float* gamma = (const float*)d_in[7];
  float* out = (float*)d_out;

  // Workspace (fp32 region then bf16 region, all 16B aligned):
  float* ws     = (float*)d_ws;
  float*  v     = ws;                       //  8,388,608 f (32 MB)
  float*  Apart = ws + 8388608;             //  4,194,304 f
  float*  A     = ws + 12582912;            //    262,144 f
  __bf16* bfb   = (__bf16*)(ws + 12845056);
  __bf16* o_bf  = bfb;                      //  8,388,608 bf16 (16 MB)
  __bf16* xv_bf = bfb + 8388608;            //  8,388,608 bf16
  __bf16* xs_bf = bfb + 16777216;           //  8,388,608 bf16
  __bf16* wo_bf = bfb + 25165824;           //     32,768 bf16
  __bf16* wv_bf = wo_bf + 32768;
  __bf16* wc_bf = wv_bf + 32768;
  // total ~97 MB

  k_prep_weights<<<128, 256, 0, stream>>>(w_o, w_v, w_c, wo_bf, wv_bf, wc_bf);
  k_proj<<<dim3(NPIX / 64, BATCH), 256, 0, stream>>>(x, wo_bf, wv_bf, b_o, b_v,
                                                     o_bf, v);
  k_softmax_colM<<<dim3(NPIX / 256, BATCH), 256, 0, stream>>>(v, xs_bf);
  k_softmax_rowN<<<BATCH * MED, 256, 0, stream>>>(v, xv_bf);
  k_gram_partial<<<dim3(16, BATCH), 256, 0, stream>>>(xv_bf, xs_bf, Apart);
  k_reduceA<<<(BATCH * MED * MED) / 256, 256, 0, stream>>>(Apart, A);
  k_final<<<dim3(NPIX / 64, BATCH), 256, 0, stream>>>(o_bf, A, wc_bf, b_c,
                                                      gamma, x, out);
}
